// DecoderLayer_19851338842283
// MI455X (gfx1250) — matrix operand order repaired and checked by reading, not compiled, not run on hardware
//
#include <hip/hip_runtime.h>

#define DEVI __device__ __forceinline__

typedef __attribute__((ext_vector_type(16))) __bf16 v16bf;
typedef __attribute__((ext_vector_type(8)))  float  v8f;

union FragAB { v16bf v; uint4 u[2]; };
union FragC  { v8f v; float f[8]; };

static constexpr int Bsz = 4, Lq = 2048, Ed = 1024, Hh = 16, Dh = 64, Fd = 4096;
static constexpr int HD = Hh * Dh;   // 1024
static constexpr int ML = Bsz * Lq;  // 8192 rows

DEVI unsigned short f2bf(float x) {
  unsigned int u = __float_as_uint(x);
  unsigned int r = u + 0x7FFFu + ((u >> 16) & 1u);
  return (unsigned short)(r >> 16);
}

// CDNA5 async copy: global -> LDS, 16 bytes per lane, tracked by ASYNCcnt.
DEVI void async_ld_b128(const void* gaddr, const void* lds_ptr) {
  unsigned lds_off = (unsigned)(uintptr_t)lds_ptr;      // addr[31:0] = LDS offset
  unsigned long long ga = (unsigned long long)(uintptr_t)gaddr;
  asm volatile("global_load_async_to_lds_b128 %0, %1, off"
               :: "v"(lds_off), "v"(ga) : "memory");
}
DEVI void wait_async_le4() { asm volatile("s_wait_asynccnt 0x4" ::: "memory"); }
DEVI void wait_async_0()   { asm volatile("s_wait_asynccnt 0x0" ::: "memory"); }

// ---------- elementwise f32 -> bf16 ----------
__global__ void cvt_bf16_kernel(const float* __restrict__ in,
                                unsigned short* __restrict__ out, int n) {
  int i = blockIdx.x * blockDim.x + threadIdx.x;
  if (i < n) out[i] = f2bf(in[i]);
}

// ---------- f32 [K][N] -> bf16 [N][K] (weight transpose) ----------
__global__ void tcvt_bf16_kernel(const float* __restrict__ in,
                                 unsigned short* __restrict__ out, int Kd, int Nd) {
  int i = blockIdx.x * blockDim.x + threadIdx.x;
  if (i < Kd * Nd) {
    int k = i / Nd, n = i - k * Nd;
    out[(size_t)n * Kd + k] = f2bf(in[i]);
  }
}

// ---------- T5 relative-position bias table: bias_tab[h][d], d = q-k >= 0 ----------
__global__ void bias_kernel(const float* __restrict__ rel_emb,
                            float* __restrict__ bias_tab) {
  int d = blockIdx.x * blockDim.x + threadIdx.x;
  if (d >= Lq) return;
  int bucket;
  if (d < 16) {
    bucket = d;
  } else {
    int v = 16 + (int)(__logf((float)d * (1.0f / 16.0f)) * (16.0f / 2.0794415417f));
    bucket = v < 31 ? v : 31;
  }
#pragma unroll
  for (int h = 0; h < Hh; ++h)
    bias_tab[h * Lq + d] = rel_emb[bucket * Hh + h];
}

// ---------- bf16 GEMM: C[M][N] = A[M][K] * Bt[N][K]^T, WMMA 16x16x32 ----------
// Double-buffered LDS fed by async global->LDS copies (ASYNCcnt pipeline).
// EPI: 0=bf16 store, 1=relu->bf16, 2=f32 store, 3=f32 + res1 + res2
template <int EPI>
__global__ __launch_bounds__(256) void gemm_bf16(
    const unsigned short* __restrict__ A, const unsigned short* __restrict__ Bt,
    int M, int N, int K, float* __restrict__ Cf, unsigned short* __restrict__ Cb,
    const float* __restrict__ res1, const float* __restrict__ res2) {
  __shared__ __align__(16) unsigned short Atile[2][128][32];
  __shared__ __align__(16) unsigned short Btile[2][128][32];

  const int t = threadIdx.x;
  const int lane = t & 31, wave = t >> 5;
  const int wm = wave >> 1, wn = wave & 1;     // 4x2 wave grid
  const int r16 = lane & 15, hi = lane >> 4;
  const int m0 = blockIdx.y * 128, n0 = blockIdx.x * 128;
  const int lrow = t >> 1, lc = (t & 1) * 16;  // 16 bf16 = 32B per thread per matrix

  const unsigned short* pArow = A + (size_t)(m0 + lrow) * K + lc;
  const unsigned short* pBrow = Bt + (size_t)(n0 + lrow) * K + lc;

  // issue one k-tile (4 x b128 async per thread)
  auto issue = [&](int kk, int buf) {
    const unsigned short* pa = pArow + kk * 32;
    const unsigned short* pb = pBrow + kk * 32;
    async_ld_b128(pa,     &Atile[buf][lrow][lc]);
    async_ld_b128(pa + 8, &Atile[buf][lrow][lc + 8]);
    async_ld_b128(pb,     &Btile[buf][lrow][lc]);
    async_ld_b128(pb + 8, &Btile[buf][lrow][lc + 8]);
  };

  FragC c[2][4];
#pragma unroll
  for (int mt = 0; mt < 2; ++mt)
#pragma unroll
    for (int nt = 0; nt < 4; ++nt)
#pragma unroll
      for (int r = 0; r < 8; ++r) c[mt][nt].f[r] = 0.f;

  const int nk = K >> 5;           // >= 32 for all our shapes
  issue(0, 0);
  issue(1, 1);

  for (int kk = 0; kk < nk; ++kk) {
    const int buf = kk & 1;
    if (kk + 1 < nk) wait_async_le4();   // 4 newer ops outstanding -> tile kk done
    else             wait_async_0();
    __syncthreads();                     // tile kk visible to all waves

    FragAB a[2], b[4];
#pragma unroll
    for (int mt = 0; mt < 2; ++mt) {
      const int ar = wm * 32 + mt * 16 + r16;
      a[mt].u[0] = *(const uint4*)&Atile[buf][ar][hi * 8];       // K 0..7 / 8..15
      a[mt].u[1] = *(const uint4*)&Atile[buf][ar][hi * 8 + 16];  // K 16..23 / 24..31
    }
#pragma unroll
    for (int nt = 0; nt < 4; ++nt) {
      const int br = wn * 64 + nt * 16 + r16;
      b[nt].u[0] = *(const uint4*)&Btile[buf][br][hi * 8];       // K 0..7 / 8..15
      b[nt].u[1] = *(const uint4*)&Btile[buf][br][hi * 8 + 16];  // K 16..23 / 24..31
    }
#pragma unroll
    for (int mt = 0; mt < 2; ++mt)
#pragma unroll
      for (int nt = 0; nt < 4; ++nt)
        c[mt][nt].v = __builtin_amdgcn_wmma_f32_16x16x32_bf16(
            false, a[mt].v, false, b[nt].v, (short)0, c[mt][nt].v, false, false);

    __syncthreads();                     // all waves done with buf before refill
    if (kk + 2 < nk) issue(kk + 2, buf);
  }

#pragma unroll
  for (int mt = 0; mt < 2; ++mt)
#pragma unroll
    for (int nt = 0; nt < 4; ++nt)
#pragma unroll
      for (int r = 0; r < 8; ++r) {
        const int m = m0 + wm * 32 + mt * 16 + r + 8 * hi;
        const int n = n0 + wn * 64 + nt * 16 + r16;
        const size_t idx = (size_t)m * N + n;
        const float v = c[mt][nt].f[r];
        if (EPI == 0) Cb[idx] = f2bf(v);
        else if (EPI == 1) Cb[idx] = f2bf(v > 0.f ? v : 0.f);
        else if (EPI == 2) Cf[idx] = v;
        else Cf[idx] = v + res1[idx] + res2[idx];
      }
}

// ---------- flash attention: per (b,h) and 64-query block ----------
__global__ __launch_bounds__(128) void attn_kernel(
    const unsigned short* __restrict__ Qb, const unsigned short* __restrict__ Kb,
    const unsigned short* __restrict__ Vb, const float* __restrict__ bias_tab,
    unsigned short* __restrict__ ctx) {
  __shared__ __align__(16) unsigned short Ks[64][64];      // [key][d]
  __shared__ __align__(16) unsigned short Vt[64][64];      // [d][key]
  __shared__ __align__(16) unsigned short Ps[4][16][64];   // per-wave P staging

  const int qb = blockIdx.x;
  const int bh = blockIdx.y;
  const int b = bh >> 4, h = bh & 15;
  const int t = threadIdx.x;
  const int lane = t & 31, wave = t >> 5;
  const int r16 = lane & 15, hi = lane >> 4;
  const int q0 = qb * 64;

  // Q A-fragments (16 rows x 64 head-dim per wave, 2 k-steps)
  const int qrow = q0 + wave * 16 + r16;
  const unsigned short* qp = Qb + ((size_t)(b * Lq + qrow)) * HD + h * Dh;
  FragAB qa[2];
#pragma unroll
  for (int ks = 0; ks < 2; ++ks) {
    qa[ks].u[0] = *(const uint4*)(qp + ks * 32 + hi * 8);
    qa[ks].u[1] = *(const uint4*)(qp + ks * 32 + hi * 8 + 16);
  }

  float m_run[8], l_run[8];
  FragC o[4];
#pragma unroll
  for (int r = 0; r < 8; ++r) { m_run[r] = -3.0e38f; l_run[r] = 0.f; }
#pragma unroll
  for (int nt = 0; nt < 4; ++nt)
#pragma unroll
    for (int r = 0; r < 8; ++r) o[nt].f[r] = 0.f;

  const int lrow = t >> 1, lc0 = (t & 1) * 32;

  for (int j = 0; j <= qb; ++j) {
    __syncthreads();   // previous block's compute done before overwrite
    {
      const size_t src = ((size_t)(b * Lq + j * 64 + lrow)) * HD + h * Dh + lc0;
      // K block: async global -> LDS (no VGPR round trip)
      const unsigned short* kp = Kb + src;
      async_ld_b128(kp,      &Ks[lrow][lc0]);
      async_ld_b128(kp + 8,  &Ks[lrow][lc0 + 8]);
      async_ld_b128(kp + 16, &Ks[lrow][lc0 + 16]);
      async_ld_b128(kp + 24, &Ks[lrow][lc0 + 24]);
      // V block: manual transpose into [d][key]
      const unsigned short* vp = Vb + src;
#pragma unroll
      for (int i = 0; i < 32; ++i) Vt[lc0 + i][lrow] = vp[i];
    }
    wait_async_0();
    __syncthreads();

    // S = Q K^T  (16x64 per wave)
    FragC s[4];
#pragma unroll
    for (int nt = 0; nt < 4; ++nt)
#pragma unroll
      for (int r = 0; r < 8; ++r) s[nt].f[r] = 0.f;
#pragma unroll
    for (int ks = 0; ks < 2; ++ks)
#pragma unroll
      for (int nt = 0; nt < 4; ++nt) {
        FragAB kb_;
        const int krow = nt * 16 + r16;
        kb_.u[0] = *(const uint4*)&Ks[krow][ks * 32 + hi * 8];
        kb_.u[1] = *(const uint4*)&Ks[krow][ks * 32 + hi * 8 + 16];
        s[nt].v = __builtin_amdgcn_wmma_f32_16x16x32_bf16(
            false, qa[ks].v, false, kb_.v, (short)0, s[nt].v, false, false);
      }

    // scale + bias + causal mask + online softmax
    const int qg_base = q0 + wave * 16 + 8 * hi;
#pragma unroll
    for (int r = 0; r < 8; ++r) {
      const int qg = qg_base + r;
      float mx = -3.0e38f;
#pragma unroll
      for (int nt = 0; nt < 4; ++nt) {
        const int kg = j * 64 + nt * 16 + r16;
        const int dd = qg - kg;
        float sv = (dd >= 0) ? s[nt].f[r] * 0.125f + bias_tab[h * Lq + dd]
                             : -3.0e38f;
        s[nt].f[r] = sv;
        mx = fmaxf(mx, sv);
      }
#pragma unroll
      for (int off = 8; off >= 1; off >>= 1)
        mx = fmaxf(mx, __shfl_xor(mx, off, 32));   // 16 lanes share one row
      const float m_new = fmaxf(m_run[r], mx);
      const float corr = __expf(m_run[r] - m_new);
      float rs = 0.f;
#pragma unroll
      for (int nt = 0; nt < 4; ++nt) {
        const float p = __expf(s[nt].f[r] - m_new);
        s[nt].f[r] = p;
        rs += p;
      }
#pragma unroll
      for (int off = 8; off >= 1; off >>= 1) rs += __shfl_xor(rs, off, 32);
      l_run[r] = l_run[r] * corr + rs;
      m_run[r] = m_new;
#pragma unroll
      for (int nt = 0; nt < 4; ++nt) o[nt].f[r] *= corr;
    }

    // stage P (C-layout -> A-layout) in wave-private LDS
#pragma unroll
    for (int nt = 0; nt < 4; ++nt)
#pragma unroll
      for (int r = 0; r < 8; ++r)
        Ps[wave][r + 8 * hi][nt * 16 + r16] = f2bf(s[nt].f[r]);
    asm volatile("s_wait_dscnt 0" ::: "memory");

    // O += P V   (K-dim = 64 keys, 2 k-steps)
#pragma unroll
    for (int ks = 0; ks < 2; ++ks) {
      FragAB pa;
      pa.u[0] = *(const uint4*)&Ps[wave][r16][ks * 32 + hi * 8];
      pa.u[1] = *(const uint4*)&Ps[wave][r16][ks * 32 + hi * 8 + 16];
#pragma unroll
      for (int nt = 0; nt < 4; ++nt) {
        FragAB vb;
        const int vr = nt * 16 + r16;
        vb.u[0] = *(const uint4*)&Vt[vr][ks * 32 + hi * 8];
        vb.u[1] = *(const uint4*)&Vt[vr][ks * 32 + hi * 8 + 16];
        o[nt].v = __builtin_amdgcn_wmma_f32_16x16x32_bf16(
            false, pa.v, false, vb.v, (short)0, o[nt].v, false, false);
      }
    }
  }

  // finalize: O /= l, store bf16 ctx in [B*L][H*D]
#pragma unroll
  for (int r = 0; r < 8; ++r) {
    const float inv = 1.0f / l_run[r];
    const int qg = q0 + wave * 16 + 8 * hi + r;
    unsigned short* cp = ctx + ((size_t)(b * Lq + qg)) * HD + h * Dh;
#pragma unroll
    for (int nt = 0; nt < 4; ++nt) cp[nt * 16 + r16] = f2bf(o[nt].f[r] * inv);
  }
}

extern "C" void kernel_launch(void* const* d_in, const int* in_sizes, int n_in,
                              void* d_out, int out_size, void* d_ws, size_t ws_size,
                              hipStream_t stream) {
  (void)in_sizes; (void)n_in; (void)out_size; (void)ws_size;
  const float* inputs  = (const float*)d_in[0];
  const float* wq      = (const float*)d_in[1];
  const float* wk      = (const float*)d_in[2];
  const float* wv      = (const float*)d_in[3];
  const float* wo      = (const float*)d_in[4];
  const float* wi      = (const float*)d_in[5];
  const float* wmo     = (const float*)d_in[6];
  const float* rel_emb = (const float*)d_in[7];
  float* out = (float*)d_out;

  char* ws = (char*)d_ws;
  size_t off = 0;
  auto alloc = [&](size_t bytes) { char* p = ws + off; off += (bytes + 255) & ~(size_t)255; return p; };
  unsigned short* Xb   = (unsigned short*)alloc((size_t)ML * Ed * 2);
  unsigned short* Qbuf = (unsigned short*)alloc((size_t)ML * HD * 2);
  unsigned short* Kbuf = (unsigned short*)alloc((size_t)ML * HD * 2);
  unsigned short* Vbuf = (unsigned short*)alloc((size_t)ML * HD * 2);
  unsigned short* Ctxb = (unsigned short*)alloc((size_t)ML * HD * 2);
  unsigned short* Hb   = (unsigned short*)alloc((size_t)ML * Fd * 2);
  float*          Attn = (float*)alloc((size_t)ML * Ed * 4);
  unsigned short* wqT  = (unsigned short*)alloc((size_t)Ed * HD * 2);
  unsigned short* wkT  = (unsigned short*)alloc((size_t)Ed * HD * 2);
  unsigned short* wvT  = (unsigned short*)alloc((size_t)Ed * HD * 2);
  unsigned short* woT  = (unsigned short*)alloc((size_t)HD * Ed * 2);
  unsigned short* wiT  = (unsigned short*)alloc((size_t)Ed * Fd * 2);
  unsigned short* wmoT = (unsigned short*)alloc((size_t)Fd * Ed * 2);
  float*          bias = (float*)alloc((size_t)Hh * Lq * 4);

  // 1) conversions
  {
    int n = ML * Ed;
    cvt_bf16_kernel<<<n / 256, 256, 0, stream>>>(inputs, Xb, n);
    tcvt_bf16_kernel<<<(Ed * HD) / 256, 256, 0, stream>>>(wq, wqT, Ed, HD);
    tcvt_bf16_kernel<<<(Ed * HD) / 256, 256, 0, stream>>>(wk, wkT, Ed, HD);
    tcvt_bf16_kernel<<<(Ed * HD) / 256, 256, 0, stream>>>(wv, wvT, Ed, HD);
    tcvt_bf16_kernel<<<(HD * Ed) / 256, 256, 0, stream>>>(wo, woT, HD, Ed);
    tcvt_bf16_kernel<<<(Ed * Fd) / 256, 256, 0, stream>>>(wi, wiT, Ed, Fd);
    tcvt_bf16_kernel<<<(Fd * Ed) / 256, 256, 0, stream>>>(wmo, wmoT, Fd, Ed);
    bias_kernel<<<Lq / 256, 256, 0, stream>>>(rel_emb, bias);
  }

  // 2) QKV projections  [8192,1024] x [1024,1024]
  dim3 gP(HD / 128, ML / 128);
  gemm_bf16<0><<<gP, 256, 0, stream>>>(Xb, wqT, ML, HD, Ed, nullptr, Qbuf, nullptr, nullptr);
  gemm_bf16<0><<<gP, 256, 0, stream>>>(Xb, wkT, ML, HD, Ed, nullptr, Kbuf, nullptr, nullptr);
  gemm_bf16<0><<<gP, 256, 0, stream>>>(Xb, wvT, ML, HD, Ed, nullptr, Vbuf, nullptr, nullptr);

  // 3) flash attention with relpos bias + causal mask
  attn_kernel<<<dim3(Lq / 64, Bsz * Hh), 128, 0, stream>>>(Qbuf, Kbuf, Vbuf, bias, Ctxb);

  // 4) output projection -> Attn (f32)
  gemm_bf16<2><<<dim3(Ed / 128, ML / 128), 256, 0, stream>>>(
      Ctxb, woT, ML, Ed, HD, Attn, nullptr, nullptr, nullptr);

  // 5) MLP: hidden = relu(X wi) bf16; out = hidden wmo + inputs + Attn
  gemm_bf16<1><<<dim3(Fd / 128, ML / 128), 256, 0, stream>>>(
      Xb, wiT, ML, Fd, Ed, nullptr, Hb, nullptr, nullptr);
  gemm_bf16<3><<<dim3(Ed / 128, ML / 128), 256, 0, stream>>>(
      Hb, wmoT, ML, Ed, Fd, out, nullptr, inputs, Attn);
}
